// RobustM2BeamLLM_19679540150605
// MI455X (gfx1250) — matrix operand order repaired and checked by reading, not compiled, not run on hardware
//
#include <hip/hip_runtime.h>
#include <hip/hip_bf16.h>

// ---------------------------------------------------------------------------
// Problem constants: B=1024, T=128, D=256, HID=512, NB=64, HH=8
// ---------------------------------------------------------------------------
#define BB   1024
#define TT   128
#define DD   256
#define HIDN 512
#define NBM  64
#define HHN  8
#define LN_EPS 1e-5f

#define SA 520   // fp32 LDS row stride (512 + 8 pad)
#define SL 72    // logits row stride

typedef __attribute__((ext_vector_type(16))) __bf16        v16bf;
typedef __attribute__((ext_vector_type(8)))  float         v8f;

// ---------------------------------------------------------------------------
// fp32 -> bf16 RNE, result in low 16 bits
// ---------------------------------------------------------------------------
__device__ __forceinline__ unsigned int f2bf(float f) {
    unsigned int u = __float_as_uint(f);
    return ((u + 0x7FFFu + ((u >> 16) & 1u)) >> 16) & 0xFFFFu;
}

// ---------------------------------------------------------------------------
// Branch-free transcendentals on native HW ops (v_tanh_f32 / v_exp_f32 / v_rcp)
// ---------------------------------------------------------------------------
__device__ __forceinline__ float rcp_(float x) {
#if __has_builtin(__builtin_amdgcn_rcpf)
    return __builtin_amdgcn_rcpf(x);
#else
    return 1.0f / x;
#endif
}
__device__ __forceinline__ float fast_tanh(float x) {
#if __has_builtin(__builtin_amdgcn_tanhf)
    return __builtin_amdgcn_tanhf(x);
#else
    const float e = __expf(2.0f * x);          // inf-safe: large x -> 1
    return 1.0f - 2.0f * rcp_(e + 1.0f);
#endif
}
__device__ __forceinline__ float fast_sigmoid(float x) {
    return rcp_(1.0f + __expf(-x));
}

// ---------------------------------------------------------------------------
// WMMA wrapper: D(16x16 f32) += A(16x32 bf16) * B(32x16 bf16)
// ---------------------------------------------------------------------------
__device__ __forceinline__ v8f wmma_bf16(v16bf a, v16bf b, v8f c) {
    return __builtin_amdgcn_wmma_f32_16x16x32_bf16(
        false, a, false, b, (short)0, c, false, false);
}

// ---------------------------------------------------------------------------
// Packed-fragment addressing.
// A 16x32 bf16 tile is stored as 512 contiguous bf16 where lane l owns 16
// contiguous elements (32B) in exact ISA A/B-fragment order.
//   element (lane,e):  lane<16: K = (e<8 ? e : 8+e),  row/N = lane&15
//                      lane>=16: K = 8 + (e<8 ? e : 8+e)
// Inverse map for producers (row, col) -> (lane, e):
//   lane = row + ((col&8)<<1),  e = (col&7) | ((col&16)>>1)   (col = K within tile)
// ---------------------------------------------------------------------------
__device__ __forceinline__ v16bf ldA_pk(const unsigned short* __restrict__ P,
                                        int kt, int lane) {
    return *(const v16bf*)(P + ((size_t)kt << 9) + (lane << 4));
}
__device__ __forceinline__ v16bf ldB(const unsigned short* __restrict__ Wp,
                                     int NT, int kt, int jt, int lane) {
    const unsigned short* p = Wp + (((size_t)(kt * NT + jt)) << 9) + (lane << 4);
    return *(const v16bf*)p;
}
__device__ __forceinline__ void stPK(unsigned short* __restrict__ P,
                                     int row, int col, float v) {
    const int kt   = col >> 5;
    const int koff = col & 31;
    const int lane = row + ((koff & 8) << 1);
    const int e    = (koff & 7) | ((koff & 16) >> 1);
    P[(kt << 9) + (lane << 4) + e] = (unsigned short)f2bf(v);
}

// ---------------------------------------------------------------------------
// Weight pre-packing: fp32 -> bf16 B-fragment layout in d_ws.
// ---------------------------------------------------------------------------
__global__ void pack_weights_kernel(const float* __restrict__ src,
                                    unsigned short* __restrict__ dst,
                                    int NT, int ld, int transpose, int total) {
    int id = blockIdx.x * blockDim.x + threadIdx.x;
    if (id >= total) return;
    int tile = id >> 9, rem = id & 511;
    int lane = rem >> 4, e = rem & 15;
    int kt = tile / NT, jt = tile % NT;
    int k = kt * 32 + ((lane < 16) ? 0 : 8) + (e & 7) + ((e & 8) << 1);
    int n = jt * 16 + (lane & 15);
    float v = transpose ? src[(size_t)n * ld + k] : src[(size_t)k * ld + n];
    dst[id] = (unsigned short)f2bf(v);
}

// ---------------------------------------------------------------------------
// GEMM: OutPK = pack( relu( A_PK @ Wp + bias ) ); 8 waves split NT col-tiles
// ---------------------------------------------------------------------------
__device__ __forceinline__ void gemm_relu_pk(const unsigned short* __restrict__ APK,
                                             const unsigned short* __restrict__ Wp,
                                             int NT, int Kt,
                                             const float* __restrict__ bias,
                                             unsigned short* __restrict__ OutPK,
                                             int wave, int lane) {
    for (int jt = wave; jt < NT; jt += 8) {
        v8f acc = {0.f,0.f,0.f,0.f,0.f,0.f,0.f,0.f};
        for (int kt = 0; kt < Kt; ++kt)
            acc = wmma_bf16(ldA_pk(APK, kt, lane), ldB(Wp, NT, kt, jt, lane), acc);
        const int col   = jt * 16 + (lane & 15);
        const int rbase = (lane < 16) ? 0 : 8;
        const float bv  = bias[col];
#pragma unroll
        for (int v = 0; v < 8; ++v)
            stPK(OutPK, v + rbase, col, fmaxf(acc[v] + bv, 0.f));
    }
}

// ---------------------------------------------------------------------------
// Main persistent kernel: one workgroup = 16 batch rows, full T scan.
// 256 threads = 8 wave32s.
// ---------------------------------------------------------------------------
__global__ __launch_bounds__(256, 1) void beam_llm_main(
    const float* __restrict__ ctx,       // (B,T,D)
    const int*   __restrict__ bh,        // (B,HH)
    const float* __restrict__ E,         // (NB,D)
    const float* __restrict__ b_in, const float* __restrict__ b_init,
    const float* __restrict__ b_ih, const float* __restrict__ b_hh,
    const float* __restrict__ ln_g, const float* __restrict__ ln_b,
    const float* __restrict__ b_o1, const float* __restrict__ b_o2,
    const unsigned short* __restrict__ WinP,   // 512x512
    const unsigned short* __restrict__ WinitP, // 512x512
    const unsigned short* __restrict__ WihTP,  // 512x1536
    const unsigned short* __restrict__ WhhTP,  // 512x1536
    const unsigned short* __restrict__ Wo1P,   // 768x512
    const unsigned short* __restrict__ Wo2P,   // 512x64
    float* __restrict__ out)             // (B,T,NB)
{
    extern __shared__ float smem[];
    float* sh_h   = smem;                    // 16*SA fp32 (GRU carry, LN input)
    float* sh_hs  = sh_h  + 16 * SA;         // 16*SA fp32 (hnew staging)
    float* sh_lg  = sh_hs + 16 * SA;         // 16*SL
    float* sh_red = sh_lg + 16 * SL;         // 544
    int*   sh_pb  = (int*)(sh_red + 544);    // 16
    unsigned short* xcatPK = (unsigned short*)(sh_pb + 16); // 16 tiles (K=512)
    unsigned short* xPK    = xcatPK + 16 * 512;             // 16 tiles (also hidden)
    unsigned short* hPK    = xPK    + 16 * 512;             // 16 tiles
    unsigned short* hcatPK = hPK    + 16 * 512;             // 24 tiles (K=768)

    const int rb   = blockIdx.x * 16;
    const int tid  = threadIdx.x;
    const int wave = tid >> 5;
    const int lane = tid & 31;

    // ---------------- init: ctx_global mean + hist_emb mean ----------------
    for (int idx = tid; idx < 16 * DD; idx += 256) {
        const int r = idx / DD, c = idx % DD;
        const float* cp = ctx + (size_t)(rb + r) * TT * DD + c;
        float s = 0.f;
        for (int t = 0; t < TT; ++t) s += cp[(size_t)t * DD];
        stPK(xcatPK, r, c, s * (1.0f / TT));

        float e = 0.f;
        const int* bhp = bh + (size_t)(rb + r) * HHN;
        for (int k = 0; k < HHN; ++k) e += E[(size_t)bhp[k] * DD + c];
        stPK(xcatPK, r, DD + c, e * (1.0f / HHN));
    }
    if (tid < 16)
        sh_pb[tid] = bh[(size_t)(rb + tid) * HHN + (HHN - 1)];
    __syncthreads();

    // h0 = tanh(xcat @ W_init + b_init)  -> sh_h (fp32) + hPK (bf16)
    for (int jt = wave; jt < 32; jt += 8) {
        v8f acc = {0.f,0.f,0.f,0.f,0.f,0.f,0.f,0.f};
        for (int kt = 0; kt < 16; ++kt)
            acc = wmma_bf16(ldA_pk(xcatPK, kt, lane),
                            ldB(WinitP, 32, kt, jt, lane), acc);
        const int col   = jt * 16 + (lane & 15);
        const int rbase = (lane < 16) ? 0 : 8;
        const float bv  = b_init[col];
#pragma unroll
        for (int v = 0; v < 8; ++v) {
            const float th = fast_tanh(acc[v] + bv);
            sh_h[(v + rbase) * SA + col] = th;
            stPK(hPK, v + rbase, col, th);
        }
    }
    __syncthreads();

    // ----------------------------- time scan -------------------------------
    for (int t = 0; t < TT; ++t) {
        // a) stage ctx_t -> xcatPK[:,0:256] + hcatPK[:,512:768]; emb -> xcatPK[:,256:512]
        for (int idx = tid; idx < 16 * DD; idx += 256) {
            const int r = idx / DD, c = idx % DD;
            const float cv = ctx[(size_t)(rb + r) * TT * DD + (size_t)t * DD + c];
            stPK(xcatPK, r, c, cv);
            stPK(hcatPK, r, HIDN + c, cv);
            const float ev = E[(size_t)sh_pb[r] * DD + c];
            stPK(xcatPK, r, DD + c, ev);
            if (t + 1 < TT)
                __builtin_prefetch(ctx + (size_t)(rb + r) * TT * DD +
                                   (size_t)(t + 1) * DD + c, 0, 3);
        }
        __syncthreads();

        // b) x = relu(xcat @ W_in + b_in) -> xPK
        gemm_relu_pk(xcatPK, WinP, 32, 16, b_in, xPK, wave, lane);
        __syncthreads();

        // c) GRU: fused 6-way gate GEMMs; hnew -> sh_hs (fp32)
        for (int jt = wave; jt < 32; jt += 8) {
            v8f ar  = {0.f,0.f,0.f,0.f,0.f,0.f,0.f,0.f};
            v8f az  = {0.f,0.f,0.f,0.f,0.f,0.f,0.f,0.f};
            v8f ain = {0.f,0.f,0.f,0.f,0.f,0.f,0.f,0.f};
            v8f ahn = {0.f,0.f,0.f,0.f,0.f,0.f,0.f,0.f};
            for (int kt = 0; kt < 16; ++kt) {
                const v16bf ax = ldA_pk(xPK, kt, lane);
                const v16bf ah = ldA_pk(hPK, kt, lane);
                ar  = wmma_bf16(ax, ldB(WihTP, 96, kt, jt,      lane), ar);
                ar  = wmma_bf16(ah, ldB(WhhTP, 96, kt, jt,      lane), ar);
                az  = wmma_bf16(ax, ldB(WihTP, 96, kt, jt + 32, lane), az);
                az  = wmma_bf16(ah, ldB(WhhTP, 96, kt, jt + 32, lane), az);
                ain = wmma_bf16(ax, ldB(WihTP, 96, kt, jt + 64, lane), ain);
                ahn = wmma_bf16(ah, ldB(WhhTP, 96, kt, jt + 64, lane), ahn);
            }
            const int col   = jt * 16 + (lane & 15);
            const int rbase = (lane < 16) ? 0 : 8;
            const float br  = b_ih[col]        + b_hh[col];
            const float bz  = b_ih[HIDN + col] + b_hh[HIDN + col];
            const float bin = b_ih[2 * HIDN + col];
            const float bhn = b_hh[2 * HIDN + col];
#pragma unroll
            for (int v = 0; v < 8; ++v) {
                const float rg = fast_sigmoid(ar[v] + br);
                const float zg = fast_sigmoid(az[v] + bz);
                const float ng = fast_tanh(ain[v] + bin + rg * (ahn[v] + bhn));
                const float ho = sh_h[(v + rbase) * SA + col];
                sh_hs[(v + rbase) * SA + col] = (1.f - zg) * ng + zg * ho;
            }
        }
        __syncthreads();

        // d) commit hnew -> sh_h (fp32) + hPK (bf16)
        for (int idx = tid; idx < 16 * HIDN; idx += 256) {
            const int r = idx >> 9, c = idx & 511;
            const float v = sh_hs[r * SA + c];
            sh_h[r * SA + c] = v;
            stPK(hPK, r, c, v);
        }
        __syncthreads();

        // e) LayerNorm(h) -> hcatPK[:,0:512]
        {
            const int r = tid >> 4, seg = tid & 15;
            float s = 0.f, q = 0.f;
            const float* hp = sh_h + r * SA + seg * 32;
#pragma unroll
            for (int i = 0; i < 32; ++i) { const float v = hp[i]; s += v; q += v * v; }
            sh_red[r * 16 + seg]       = s;
            sh_red[256 + r * 16 + seg] = q;
        }
        __syncthreads();
        if (tid < 16) {
            float s = 0.f, q = 0.f;
#pragma unroll
            for (int i = 0; i < 16; ++i) {
                s += sh_red[tid * 16 + i];
                q += sh_red[256 + tid * 16 + i];
            }
            const float mu  = s * (1.0f / HIDN);
            const float var = q * (1.0f / HIDN) - mu * mu;
            sh_red[512 + tid] = mu;
            sh_red[528 + tid] = rsqrtf(var + LN_EPS);
        }
        __syncthreads();
        for (int idx = tid; idx < 16 * HIDN; idx += 256) {
            const int r = idx >> 9, c = idx & 511;
            const float hn = (sh_h[r * SA + c] - sh_red[512 + r]) *
                             sh_red[528 + r] * ln_g[c] + ln_b[c];
            stPK(hcatPK, r, c, hn);
        }
        __syncthreads();

        // f) hidden = relu(hcat @ W_o1 + b_o1) -> xPK (reuse)
        gemm_relu_pk(hcatPK, Wo1P, 32, 24, b_o1, xPK, wave, lane);
        __syncthreads();

        // g) logits = hidden @ W_o2 + b_o2 (waves 0..3); LDS + global store
        if (wave < 4) {
            const int jt = wave;
            v8f acc = {0.f,0.f,0.f,0.f,0.f,0.f,0.f,0.f};
            for (int kt = 0; kt < 16; ++kt)
                acc = wmma_bf16(ldA_pk(xPK, kt, lane),
                                ldB(Wo2P, 4, kt, jt, lane), acc);
            const int col   = jt * 16 + (lane & 15);
            const int rbase = (lane < 16) ? 0 : 8;
            const float bv  = b_o2[col];
#pragma unroll
            for (int v = 0; v < 8; ++v) {
                const float val = acc[v] + bv;
                sh_lg[(v + rbase) * SL + col] = val;
                out[(size_t)(rb + v + rbase) * TT * NBM + (size_t)t * NBM + col] = val;
            }
        }
        __syncthreads();

        // h) argmax per row (first max, matching jnp.argmax)
        if (tid < 16) {
            const float* lp = sh_lg + tid * SL;
            float best = lp[0];
            int   bi   = 0;
            for (int n = 1; n < NBM; ++n)
                if (lp[n] > best) { best = lp[n]; bi = n; }
            sh_pb[tid] = bi;
        }
        __syncthreads();
    }
}

// ---------------------------------------------------------------------------
// Host side
// ---------------------------------------------------------------------------
extern "C" void kernel_launch(void* const* d_in, const int* in_sizes, int n_in,
                              void* d_out, int out_size, void* d_ws, size_t ws_size,
                              hipStream_t stream) {
    (void)in_sizes; (void)n_in; (void)out_size; (void)ws_size;

    const float* ctx    = (const float*)d_in[0];
    const int*   bh     = (const int*)  d_in[1];
    const float* E      = (const float*)d_in[2];
    const float* W_in   = (const float*)d_in[3];
    const float* b_in   = (const float*)d_in[4];
    const float* W_init = (const float*)d_in[5];
    const float* b_init = (const float*)d_in[6];
    const float* W_ih   = (const float*)d_in[7];
    const float* W_hh   = (const float*)d_in[8];
    const float* b_ih   = (const float*)d_in[9];
    const float* b_hh   = (const float*)d_in[10];
    const float* ln_g   = (const float*)d_in[11];
    const float* ln_b   = (const float*)d_in[12];
    const float* W_o1   = (const float*)d_in[13];
    const float* b_o1   = (const float*)d_in[14];
    const float* W_o2   = (const float*)d_in[15];
    const float* b_o2   = (const float*)d_in[16];
    float* out = (float*)d_out;

    const size_t OFF_WIN   = 0;
    const size_t OFF_WINIT = OFF_WIN   + 512ull * 512 * 2;
    const size_t OFF_WIHT  = OFF_WINIT + 512ull * 512 * 2;
    const size_t OFF_WHHT  = OFF_WIHT  + 512ull * 1536 * 2;
    const size_t OFF_WO1   = OFF_WHHT  + 512ull * 1536 * 2;
    const size_t OFF_WO2   = OFF_WO1   + 768ull * 512 * 2;

    auto wsp = [&](size_t off) { return (unsigned short*)((char*)d_ws + off); };
    auto pack = [&](const float* w, size_t off, int Ktiles, int NT, int ld, int tr) {
        const int total = Ktiles * NT * 512;
        pack_weights_kernel<<<(total + 255) / 256, 256, 0, stream>>>(
            w, wsp(off), NT, ld, tr, total);
    };

    pack(W_in,   OFF_WIN,   16, 32, 512, 0);
    pack(W_init, OFF_WINIT, 16, 32, 512, 0);
    pack(W_ih,   OFF_WIHT,  16, 96, 512, 1);   // W_ih.T : 512 x 1536
    pack(W_hh,   OFF_WHHT,  16, 96, 512, 1);
    pack(W_o1,   OFF_WO1,   24, 32, 512, 0);
    pack(W_o2,   OFF_WO2,   16, 4,  64,  0);

    const size_t smem_bytes =
        (size_t)(2 * 16 * SA + 16 * SL + 544) * sizeof(float) + 16 * sizeof(int) +
        (size_t)(3 * 16 * 512 + 24 * 512) * sizeof(unsigned short);
    (void)hipFuncSetAttribute((const void*)beam_llm_main,
                              hipFuncAttributeMaxDynamicSharedMemorySize,
                              (int)smem_bytes);

    beam_llm_main<<<BB / 16, 256, smem_bytes, stream>>>(
        ctx, bh, E, b_in, b_init, b_ih, b_hh, ln_g, ln_b, b_o1, b_o2,
        wsp(OFF_WIN), wsp(OFF_WINIT), wsp(OFF_WIHT), wsp(OFF_WHHT),
        wsp(OFF_WO1), wsp(OFF_WO2), out);
}
